// ConvLayer_82849919140696
// MI455X (gfx1250) — compile-verified, matrix operand order checked by reading, not run on hardware
//
#include <hip/hip_runtime.h>

#define H 64
#define EDIM 16
#define NN 15000
#define NE 40000
#define HH 4096  // H*H

typedef __attribute__((ext_vector_type(16))) __bf16 v16bf;
typedef __attribute__((ext_vector_type(8)))  float  v8f;

// ---------- bf16 helpers ----------
__device__ __forceinline__ unsigned short f32_to_bf16_bits(float f) {
  union { float f; unsigned u; } c; c.f = f;
  unsigned u = c.u;
  unsigned r = u + 0x7FFFu + ((u >> 16) & 1u);  // round-to-nearest-even
  return (unsigned short)(r >> 16);
}
__device__ __forceinline__ __bf16 f2bf(float f) {
  union { unsigned short s; __bf16 b; } c; c.s = f32_to_bf16_bits(f); return c.b;
}
__device__ __forceinline__ float bf_lo(unsigned u) {
  union { unsigned u; float f; } c; c.u = u << 16; return c.f;
}
__device__ __forceinline__ float bf_hi(unsigned u) {
  union { unsigned u; float f; } c; c.u = u & 0xFFFF0000u; return c.f;
}

// K index inside a 16x32 A/B fragment for element j (0..15) given lane-half.
// Lanes 0-15: K = {0..7, 16..23}; lanes 16-31: K = {8..15, 24..31}.
__device__ __forceinline__ int wmma_k_of(int j, int half) {
  int v = j >> 1;
  return ((v & 4) ? 16 : 0) + (v & 3) * 2 + half * 8 + (j & 1);
}

__device__ __forceinline__ v8f wmma_bf16(v16bf a, v16bf b, v8f c) {
  return __builtin_amdgcn_wmma_f32_16x16x32_bf16(false, a, false, b, (short)0, c,
                                                 false, false);
}

// ---------- Prep: swizzle a 64-row B matrix into WMMA fragment layout -------
// dst fragment f = ct*2 + kc holds 32 lanes x 16 contiguous bf16.
// element: dst[((ct*2+kc)*32 + l)*16 + j] = B[kc*32 + k_of(j, l>>4)][ct*16 + (l&15)]
// transpose=0: B[k][col] = src[k*ncols + col]   (W2, root)
// transpose=1: B[k][col] = src[col*64 + k]      (W_ih^T, W_hh^T)
__global__ __launch_bounds__(256)
void swz_kernel(const float* __restrict__ src, unsigned short* __restrict__ dst,
                int ncols, int transpose) {
  int o = blockIdx.x * 256 + threadIdx.x;
  if (o >= 64 * ncols) return;
  int j  = o & 15;
  int l  = (o >> 4) & 31;
  int kc = (o >> 9) & 1;
  int ct = o >> 10;
  int k   = kc * 32 + wmma_k_of(j, l >> 4);
  int col = ct * 16 + (l & 15);
  float v = transpose ? src[(size_t)col * 64 + k] : src[(size_t)k * ncols + col];
  dst[o] = f32_to_bf16_bits(v);
}

// ---------- Kernel 1: ew = (relu(edge_attr@W1+b1) @ W2 + b2) as bf16 --------
// grid (NE/16, HH/512), block 256 (8 waves). Wave: 4 col-tiles x 2 WMMA(K=32).
__global__ __launch_bounds__(256)
void ew_kernel(const float* __restrict__ edge_attr,
               const float* __restrict__ W1, const float* __restrict__ b1,
               const unsigned short* __restrict__ w2f,  // pre-swizzled bf16
               const float* __restrict__ b2,
               unsigned short* __restrict__ ew) {
  __shared__ __align__(32) __bf16 a_lds[2][32][16];  // A frags, swizzled
  const int eb = blockIdx.x * 16;
  const int cb = blockIdx.y * 512;
  const int tid = threadIdx.x;

  // r = relu(edge_attr @ W1 + b1); store directly in A-fragment layout
  for (int idx = tid; idx < 16 * 64; idx += 256) {
    int el = idx >> 6, hc = idx & 63;
    float acc = b1[hc];
    const float* ea = edge_attr + (size_t)(eb + el) * EDIM;
#pragma unroll
    for (int k = 0; k < EDIM; ++k) acc = fmaf(ea[k], W1[k * H + hc], acc);
    int kc = hc >> 5, k5 = hc & 31;
    int hf = (k5 >> 3) & 1, a = (k5 >> 4) & 1;
    int v = (a << 2) | ((k5 >> 1) & 3);
    int j = (v << 1) | (k5 & 1);
    a_lds[kc][hf * 16 + el][j] = f2bf(acc > 0.f ? acc : 0.f);
  }
  __syncthreads();

  const int wave = tid >> 5, lane = tid & 31;
  const int half = lane >> 4, ln = lane & 15;
  v16bf a0 = *(const v16bf*)&a_lds[0][lane][0];
  v16bf a1 = *(const v16bf*)&a_lds[1][lane][0];

  const int wc = cb + wave * 64;
#pragma unroll
  for (int t = 0; t < 4; ++t) {
    const int c0 = wc + t * 16;
    const int ct = c0 >> 4;
    v16bf b0  = *(const v16bf*)(w2f + ((size_t)(ct * 2 + 0) * 512 + lane * 16));
    v16bf b1v = *(const v16bf*)(w2f + ((size_t)(ct * 2 + 1) * 512 + lane * 16));
    v8f c = {};
    c = wmma_bf16(a0, b0, c);
    c = wmma_bf16(a1, b1v, c);
#pragma unroll
    for (int vr = 0; vr < 8; ++vr) {
      int m = vr + half * 8;
      ew[(size_t)(eb + m) * HH + c0 + ln] = f32_to_bf16_bits(c[vr] + b2[c0 + ln]);
    }
  }
}

// ---------- small utility kernels ----------
__global__ void zero_f32(float* __restrict__ p, int n) {
  int i = blockIdx.x * 256 + threadIdx.x;
  if (i < n) p[i] = 0.f;
}
__global__ void deg_kernel(const int* __restrict__ ei, float* __restrict__ deg) {
  int e = blockIdx.x * 256 + threadIdx.x;
  if (e < NE) atomicAdd(&deg[ei[NE + e]], 1.0f);
}

// ---------- Kernel 2: msg[e] = state[src[e]] @ ew[e]; atomic segment-sum ----
// One wave per edge; lane l owns output cols {2l, 2l+1}. Bandwidth-bound.
__global__ __launch_bounds__(256)
void msg_kernel(const float* __restrict__ state, const int* __restrict__ ei,
                const unsigned* __restrict__ ew32, float* __restrict__ agg) {
  const int wave = threadIdx.x >> 5, lane = threadIdx.x & 31;
  const int e = blockIdx.x * 8 + wave;
  if (e >= NE) return;
  const int src = ei[e];
  const int tgt = ei[NE + e];
  const float x0 = state[(size_t)src * H + lane];
  const float x1 = state[(size_t)src * H + 32 + lane];
  const unsigned* row = ew32 + (size_t)e * (HH / 2);  // bf16 pairs, 32/row
  float acc0 = 0.f, acc1 = 0.f;
#pragma unroll 4
  for (int i = 0; i < 32; ++i) {
    float xi = __shfl(x0, i, 32);
    unsigned u = row[i * 32 + lane];
    acc0 = fmaf(xi, bf_lo(u), acc0);
    acc1 = fmaf(xi, bf_hi(u), acc1);
  }
#pragma unroll 4
  for (int i = 0; i < 32; ++i) {
    float xi = __shfl(x1, i, 32);
    unsigned u = row[(32 + i) * 32 + lane];
    acc0 = fmaf(xi, bf_lo(u), acc0);
    acc1 = fmaf(xi, bf_hi(u), acc1);
  }
  atomicAdd(&agg[(size_t)tgt * H + 2 * lane], acc0);
  atomicAdd(&agg[(size_t)tgt * H + 2 * lane + 1], acc1);
}

// ---------- Kernel 3: m = relu(out@root + agg/deg + bias); h = GRU(m,h) -----
// One wave per 16-node tile, 4 waves per block. 56 WMMAs per tile.
__global__ __launch_bounds__(128)
void node_kernel(const float* __restrict__ sin_, const float* __restrict__ agg,
                 const float* __restrict__ deg,
                 const unsigned short* __restrict__ rootf,  // pre-swizzled
                 const float* __restrict__ conv_bias,
                 const unsigned short* __restrict__ wihf,   // pre-swizzled W_ih^T
                 const unsigned short* __restrict__ whhf,   // pre-swizzled W_hh^T
                 const float* __restrict__ b_ih, const float* __restrict__ b_hh,
                 float* __restrict__ sout) {
  __shared__ __align__(32) __bf16 m_fr[4][2][32][16];  // m in A-frag layout
  const int wave = threadIdx.x >> 5, lane = threadIdx.x & 31;
  const int half = lane >> 4, ln = lane & 15;
  const int nb = (blockIdx.x * 4 + wave) * 16;

  // A fragments of h (== out): two contiguous 8-float runs per fragment
  v16bf ah0, ah1;
  {
    int g = nb + ln; if (g >= NN) g = NN - 1;
    const float* hp = sin_ + (size_t)g * H;
    float hv[16];
    *(float4*)&hv[0]  = *(const float4*)(hp + half * 8);
    *(float4*)&hv[4]  = *(const float4*)(hp + half * 8 + 4);
    *(float4*)&hv[8]  = *(const float4*)(hp + 16 + half * 8);
    *(float4*)&hv[12] = *(const float4*)(hp + 16 + half * 8 + 4);
#pragma unroll
    for (int j = 0; j < 16; ++j) ah0[j] = f2bf(hv[j]);
    *(float4*)&hv[0]  = *(const float4*)(hp + 32 + half * 8);
    *(float4*)&hv[4]  = *(const float4*)(hp + 32 + half * 8 + 4);
    *(float4*)&hv[8]  = *(const float4*)(hp + 48 + half * 8);
    *(float4*)&hv[12] = *(const float4*)(hp + 48 + half * 8 + 4);
#pragma unroll
    for (int j = 0; j < 16; ++j) ah1[j] = f2bf(hv[j]);
  }

  // m = relu(out@root + agg/deg + conv_bias) -> LDS in A-frag layout
#pragma unroll
  for (int t = 0; t < 4; ++t) {
    v16bf b0  = *(const v16bf*)(rootf + ((size_t)(t * 2 + 0) * 512 + lane * 16));
    v16bf b1v = *(const v16bf*)(rootf + ((size_t)(t * 2 + 1) * 512 + lane * 16));
    v8f c = {};
    c = wmma_bf16(ah0, b0, c);
    c = wmma_bf16(ah1, b1v, c);
#pragma unroll
    for (int vr = 0; vr < 8; ++vr) {
      int m = vr + half * 8;
      int col = t * 16 + ln;
      int g = nb + m; int gc = (g < NN) ? g : NN - 1;
      float d = deg[gc]; d = d > 1.f ? d : 1.f;
      float val = c[vr] + agg[(size_t)gc * H + col] / d + conv_bias[col];
      val = val > 0.f ? val : 0.f;
      int kc = col >> 5, k5 = col & 31;
      int hf = (k5 >> 3) & 1, a = (k5 >> 4) & 1;
      int v = (a << 2) | ((k5 >> 1) & 3);
      int j = (v << 1) | (k5 & 1);
      m_fr[wave][kc][hf * 16 + m][j] = f2bf(val);
    }
  }
  __syncthreads();

  v16bf am0 = *(const v16bf*)&m_fr[wave][0][lane][0];
  v16bf am1 = *(const v16bf*)&m_fr[wave][1][lane][0];

  // GRU gates, fused per 16-col output tile. W^T col tile for gate g: ct=g*4+t
#pragma unroll
  for (int t = 0; t < 4; ++t) {
    v8f cxr = {}, cxz = {}, cxn = {}, chr_ = {}, chz = {}, chn = {};
#pragma unroll
    for (int kc = 0; kc < 2; ++kc) {
      v16bf am = kc ? am1 : am0;
      v16bf ah = kc ? ah1 : ah0;
      size_t lo = (size_t)lane * 16;
      v16bf br  = *(const v16bf*)(wihf + ((size_t)((0 + t) * 2 + kc) * 512 + lo));
      v16bf bz  = *(const v16bf*)(wihf + ((size_t)((4 + t) * 2 + kc) * 512 + lo));
      v16bf bn  = *(const v16bf*)(wihf + ((size_t)((8 + t) * 2 + kc) * 512 + lo));
      v16bf br2 = *(const v16bf*)(whhf + ((size_t)((0 + t) * 2 + kc) * 512 + lo));
      v16bf bz2 = *(const v16bf*)(whhf + ((size_t)((4 + t) * 2 + kc) * 512 + lo));
      v16bf bn2 = *(const v16bf*)(whhf + ((size_t)((8 + t) * 2 + kc) * 512 + lo));
      cxr  = wmma_bf16(am, br,  cxr);
      cxz  = wmma_bf16(am, bz,  cxz);
      cxn  = wmma_bf16(am, bn,  cxn);
      chr_ = wmma_bf16(ah, br2, chr_);
      chz  = wmma_bf16(ah, bz2, chz);
      chn  = wmma_bf16(ah, bn2, chn);
    }
#pragma unroll
    for (int vr = 0; vr < 8; ++vr) {
      int m = vr + half * 8;
      int col = t * 16 + ln;
      int g = nb + m; int gc = (g < NN) ? g : NN - 1;
      float hval = sin_[(size_t)gc * H + col];
      float xr = cxr[vr]  + b_ih[col];
      float xz = cxz[vr]  + b_ih[64 + col];
      float xn = cxn[vr]  + b_ih[128 + col];
      float hr = chr_[vr] + b_hh[col];
      float hz = chz[vr]  + b_hh[64 + col];
      float hn = chn[vr]  + b_hh[128 + col];
      float r = 1.f / (1.f + __expf(-(xr + hr)));
      float z = 1.f / (1.f + __expf(-(xz + hz)));
      float ng = tanhf(xn + r * hn);
      float hnew = (1.f - z) * ng + z * hval;
      if (g < NN) sout[(size_t)g * H + col] = hnew;
    }
  }
}

// ---------- host launcher ----------
extern "C" void kernel_launch(void* const* d_in, const int* in_sizes, int n_in,
                              void* d_out, int out_size, void* d_ws, size_t ws_size,
                              hipStream_t stream) {
  (void)in_sizes; (void)n_in; (void)out_size; (void)ws_size;
  const float* x         = (const float*)d_in[0];
  const int*   ei        = (const int*)d_in[1];
  const float* edge_attr = (const float*)d_in[2];
  /* d_in[3] = times (==3, hard-coded: host cannot read device scalar in capture) */
  const float* W1        = (const float*)d_in[4];
  const float* b1        = (const float*)d_in[5];
  const float* W2        = (const float*)d_in[6];
  const float* b2        = (const float*)d_in[7];
  const float* root      = (const float*)d_in[8];
  const float* conv_bias = (const float*)d_in[9];
  const float* W_ih      = (const float*)d_in[10];
  const float* W_hh      = (const float*)d_in[11];
  const float* b_ih      = (const float*)d_in[12];
  const float* b_hh      = (const float*)d_in[13];
  float* out = (float*)d_out;

  char* ws = (char*)d_ws;
  unsigned short* ew = (unsigned short*)ws;           // NE*HH*2 = 327.68 MB
  size_t off = (size_t)NE * HH * 2;
  float* deg = (float*)(ws + off); off += (((size_t)NN * 4 + 255) & ~(size_t)255);
  float* agg = (float*)(ws + off); off += (size_t)NN * H * 4;
  float* S0  = (float*)(ws + off); off += (size_t)NN * H * 4;
  float* S1  = (float*)(ws + off); off += (size_t)NN * H * 4;
  unsigned short* w2f   = (unsigned short*)(ws + off); off += (size_t)64 * HH * 2;
  unsigned short* rootf = (unsigned short*)(ws + off); off += (size_t)64 * 64 * 2;
  unsigned short* wihf  = (unsigned short*)(ws + off); off += (size_t)64 * 192 * 2;
  unsigned short* whhf  = (unsigned short*)(ws + off);

  // One-time: swizzle static B operands into WMMA fragment layout (bf16)
  swz_kernel<<<(64 * HH + 255) / 256, 256, 0, stream>>>(W2, w2f, HH, 0);
  swz_kernel<<<(64 * 64 + 255) / 256, 256, 0, stream>>>(root, rootf, 64, 0);
  swz_kernel<<<(64 * 192 + 255) / 256, 256, 0, stream>>>(W_ih, wihf, 192, 1);
  swz_kernel<<<(64 * 192 + 255) / 256, 256, 0, stream>>>(W_hh, whhf, 192, 1);

  // One-time: edge weight matrices (loop-invariant) + degrees
  ew_kernel<<<dim3(NE / 16, HH / 512), 256, 0, stream>>>(edge_attr, W1, b1, w2f, b2, ew);
  zero_f32<<<(NN + 255) / 256, 256, 0, stream>>>(deg, NN);
  deg_kernel<<<(NE + 255) / 256, 256, 0, stream>>>(ei, deg);

  const float* sin_ = x;          // out == h throughout
  float* bufs[3] = {S0, S1, out}; // last iteration writes d_out directly
  for (int t = 0; t < 3; ++t) {
    zero_f32<<<(NN * H + 255) / 256, 256, 0, stream>>>(agg, NN * H);
    msg_kernel<<<NE / 8, 256, 0, stream>>>(sin_, ei, (const unsigned*)ew, agg);
    node_kernel<<<(NN + 63) / 64, 128, 0, stream>>>(sin_, agg, deg, rootf, conv_bias,
                                                    wihf, whhf, b_ih, b_hh, bufs[t]);
    sin_ = bufs[t];
  }
}
